// JimmyMarkIV_59837484368464
// MI455X (gfx1250) — matrix operand order repaired and checked by reading, not compiled, not run on hardware
//
#include <hip/hip_runtime.h>
#include <hip/hip_bf16.h>

#define IN_DIM 2048
#define T_UNITS 8192
#define OUT_DIM 1024
#define NPROP 8
#define KSPLIT 8
#define MT 128   // build tile dim

typedef __attribute__((ext_vector_type(16))) __bf16 v16bf;
typedef __attribute__((ext_vector_type(8)))  float  v8f;

union TileBits {
    v16bf v;
    uint4 q[2];
};

// ---------------------------------------------------------------------------
// Build dense symmetric bf16 matrix M from flat strict-upper-triangle weights,
// tiled through LDS so that BOTH halves get coalesced global traffic:
//   - read w_flat coalesced for the upper-triangle tile (contiguous j),
//   - write the tile and its transpose coalesced (LDS does the transpose).
// bf16 M is 128 MB -> resident in the 192 MB L2 for all 8 matvec steps.
// ---------------------------------------------------------------------------
__global__ void __launch_bounds__(256)
build_M_tiled_kernel(const float* __restrict__ w,
                     __hip_bfloat16* __restrict__ M) {
    const int bi = blockIdx.y;   // i-block
    const int bj = blockIdx.x;   // j-block
    if (bj < bi) return;         // upper-triangle tiles only

    __shared__ __hip_bfloat16 tile[MT][MT + 2];  // +2 elems: rotate LDS banks

    const int i0 = bi * MT, j0 = bj * MT;

    // Load + convert. For off-diagonal tiles (the vast majority) gi < gj
    // always, so w reads are contiguous in j -> fully coalesced.
    for (int e = threadIdx.x; e < MT * MT; e += 256) {
        const int r = e >> 7;
        const int c = e & (MT - 1);
        const int gi = i0 + r, gj = j0 + c;
        float v = 0.0f;
        if (gi != gj) {
            const int ii = (gi < gj) ? gi : gj;
            const int jj = (gi < gj) ? gj : gi;
            const size_t base = (size_t)ii * (size_t)(2 * T_UNITS - ii - 1) / 2;
            v = w[base + (size_t)(jj - ii - 1)];
        }
        tile[r][c] = __float2bfloat16(v);
    }
    __syncthreads();

    // Upper block: M[i0+r][j0+c] — coalesced (contiguous c).
    for (int e = threadIdx.x; e < MT * MT; e += 256) {
        const int r = e >> 7, c = e & (MT - 1);
        M[(size_t)(i0 + r) * T_UNITS + (j0 + c)] = tile[r][c];
    }
    // Mirrored block: M[j0+r][i0+c] = tile[c][r] — coalesced global store,
    // transposed LDS read (padding avoids systematic bank conflicts).
    if (bi != bj) {
        for (int e = threadIdx.x; e < MT * MT; e += 256) {
            const int r = e >> 7, c = e & (MT - 1);
            M[(size_t)(j0 + r) * T_UNITS + (i0 + c)] = tile[c][r];
        }
    }
}

// ---------------------------------------------------------------------------
// s0[:IN_DIM] = inputs, rest 0.  Keep both f32 (final output) and bf16 (WMMA).
// ---------------------------------------------------------------------------
__global__ void init_state_kernel(const float* __restrict__ inp,
                                  float* __restrict__ s_f32,
                                  __hip_bfloat16* __restrict__ s_bf16) {
    const int t = blockIdx.x * blockDim.x + threadIdx.x;
    const float v = (t < IN_DIM) ? inp[t] : 0.0f;
    s_f32[t]  = v;
    s_bf16[t] = __float2bfloat16(v);
}

// ---------------------------------------------------------------------------
// One K-slice of the matvec via v_wmma_f32_16x16x32_bf16.
// Each wave: 16 output rows (one strip), K range of T/KSPLIT (32 tiles).
// B operand = s chunk replicated across all 16 columns, so every column of D
// equals M_tile @ s; lane 0 holds rows 0..7, lane 16 holds rows 8..15.
// ---------------------------------------------------------------------------
__global__ void __launch_bounds__(256)
matvec_wmma_kernel(const __hip_bfloat16* __restrict__ M,
                   const __hip_bfloat16* __restrict__ s,
                   float* __restrict__ partials) {
    const int lane  = threadIdx.x & 31;
    const int wave  = threadIdx.x >> 5;
    const int strip = blockIdx.x * 8 + wave;      // 0..511
    const int r0    = strip * 16;
    const int half  = lane >> 4;                  // 0 or 1
    const int lr    = lane & 15;                  // row within strip

    const int kBegin = blockIdx.y * (T_UNITS / KSPLIT);
    const int kEnd   = kBegin + (T_UNITS / KSPLIT);

    const __hip_bfloat16* __restrict__ Arow = M + (size_t)(r0 + lr) * T_UNITS;

    v8f c = {};
#pragma unroll 4
    for (int k = kBegin; k < kEnd; k += 32) {
        TileBits a, b;
        // A (16x32 bf16): lanes 0-15 carry K {k..k+7, k+16..k+23},
        //                 lanes 16-31 carry K {k+8..k+15, k+24..k+31}
        a.q[0] = *(const uint4*)(Arow + k + half * 8);
        a.q[1] = *(const uint4*)(Arow + k + 16 + half * 8);
        // B (32x16 bf16): lanes 0-15 carry K {k..k+7, k+16..k+23}, lanes 16-31
        // carry K {k+8..k+15, k+24..k+31}; identical data per lane-half => replicated columns.
        const __hip_bfloat16* sp = s + k + half * 8;
        b.q[0] = *(const uint4*)(sp);
        b.q[1] = *(const uint4*)(sp + 16);

        c = __builtin_amdgcn_wmma_f32_16x16x32_bf16(
            /*neg_a=*/false, a.v, /*neg_b=*/false, b.v,
            /*c_mod=*/(short)0, c, /*reuse_a=*/false, /*reuse_b=*/false);
    }

    float* __restrict__ outp = partials + (size_t)blockIdx.y * T_UNITS + r0;
    if (lane == 0) {
#pragma unroll
        for (int p = 0; p < 8; ++p) outp[p] = c[p];
    } else if (lane == 16) {
#pragma unroll
        for (int p = 0; p < 8; ++p) outp[8 + p] = c[p];
    }
}

// ---------------------------------------------------------------------------
// s_new[t] = tanh(sum_k partials[k][t] + b[t]); emit f32 + bf16 copies.
// Deterministic fixed-order sum over the KSPLIT partial buffers.
// ---------------------------------------------------------------------------
__global__ void reduce_tanh_kernel(const float* __restrict__ partials,
                                   const float* __restrict__ bias,
                                   float* __restrict__ s_f32,
                                   __hip_bfloat16* __restrict__ s_bf16) {
    const int t = blockIdx.x * blockDim.x + threadIdx.x;
    float acc = bias[t];
#pragma unroll
    for (int ks = 0; ks < KSPLIT; ++ks)
        acc += partials[(size_t)ks * T_UNITS + t];
    const float v = tanhf(acc);
    s_f32[t]  = v;
    s_bf16[t] = __float2bfloat16(v);
}

__global__ void copy_out_kernel(const float* __restrict__ s_f32,
                                float* __restrict__ out) {
    const int t = blockIdx.x * blockDim.x + threadIdx.x;
    out[t] = s_f32[T_UNITS - OUT_DIM + t];
}

// ---------------------------------------------------------------------------
// Workspace layout (bytes):
//   [0)        M bf16    : 8192*8192*2   = 134217728
//   [M_end)    s_f32     : 8192*4        = 32768
//   [..)       s_bf16    : 8192*2        = 16384
//   [..)       partials  : KSPLIT*8192*4 = 262144
// ---------------------------------------------------------------------------
extern "C" void kernel_launch(void* const* d_in, const int* in_sizes, int n_in,
                              void* d_out, int out_size, void* d_ws, size_t ws_size,
                              hipStream_t stream) {
    const float* d_inputs = (const float*)d_in[0];
    const float* d_wflat  = (const float*)d_in[1];
    const float* d_bias   = (const float*)d_in[2];
    float* d_output       = (float*)d_out;

    char* ws = (char*)d_ws;
    const size_t M_bytes = (size_t)T_UNITS * T_UNITS * sizeof(__hip_bfloat16);
    __hip_bfloat16* d_M     = (__hip_bfloat16*)ws;
    float*          d_sf32  = (float*)(ws + M_bytes);
    __hip_bfloat16* d_sbf16 = (__hip_bfloat16*)(ws + M_bytes + T_UNITS * sizeof(float));
    float*          d_part  = (float*)(ws + M_bytes + T_UNITS * sizeof(float)
                                          + T_UNITS * sizeof(__hip_bfloat16));

    // 1) Materialize symmetric bf16 matrix (one time, all-coalesced, then
    //    L2-resident for the matvec steps).
    build_M_tiled_kernel<<<dim3(T_UNITS / MT, T_UNITS / MT), 256, 0, stream>>>(
        d_wflat, d_M);

    // 2) Initialize state.
    init_state_kernel<<<T_UNITS / 256, 256, 0, stream>>>(d_inputs, d_sf32, d_sbf16);

    // 3) 8 propagation steps: WMMA matvec (K split into 8 slices -> 4096
    //    waves for latency hiding) + tanh reduce.
    for (int step = 0; step < NPROP; ++step) {
        matvec_wmma_kernel<<<dim3(T_UNITS / 16 / 8, KSPLIT), 256, 0, stream>>>(
            d_M, d_sbf16, d_part);
        reduce_tanh_kernel<<<T_UNITS / 256, 256, 0, stream>>>(
            d_part, d_bias, d_sf32, d_sbf16);
    }

    // 4) Emit last OUT_DIM states.
    copy_out_kernel<<<OUT_DIM / 256, 256, 0, stream>>>(d_sf32, d_output);
}
